// LearnableQuantization_71287867179494
// MI455X (gfx1250) — compile-verified, matrix operand order checked by reading, not run on hardware
//
#include <hip/hip_runtime.h>
#include <hip/hip_bf16.h>

// LearnableQuantization forward for MI455X (gfx1250, wave32).
//
// Math (per element), all in log2 domain:
//   t2_k = a2*(k-8.5) + b2            (a2 = (alpha/dev)*log2e, b2 = -(x/dev)*log2e)
//   s_k  = softplus2(-t2_k) = max(-t2,0) + log2(1+2^-| -t2 |)   [= softplus(t)/ln2]
//   num2_k = s_k - s_{k+1} + 0.2*log2e                          [= num_k/ln2]
//   nlr2_k = -log2(r_k)                                         [= (-ln r_k)/ln2]
//   w_k  = sqrt(num2_k)*rsqrt(nlr2_k)   == sqrt(num_k)*rsqrt(-ln r_k)  (ln2 cancels)
//   out  = alpha * (sum w_k*(k-8.5)) / (sum w_k)   [softmax denominator cancels]
//
// Wave layout: lane L in [0,16) handles bins 0-7 of element L, lane L+16 handles
// bins 8-15 of element L -- exactly the B-matrix layout of V_WMMA_F32_16X16X32_F16
// (lanes 0-15: K=0..7, lanes 16-31: K=8..15, K>=16 zero). A row0 = grid base
// values, row1 = ones -> one WMMA yields both 16-bin reductions for 16 elements
// with zero cross-lane shuffles.

typedef _Float16 v16h __attribute__((ext_vector_type(16)));
typedef float    v8f  __attribute__((ext_vector_type(8)));

#define LOG2E_F    1.4426950408889634f
#define NOISE2_C   0.2885390081777927f   // 0.2 * log2(e)

__device__ __forceinline__ float neg_log2_urand(unsigned s) {
    // short mix -> uniform (0,1) -> -log2(r) > 0   (single v_log_f32)
    s *= 0x85ebca6bu; s ^= s >> 13;
    s *= 0xc2b2ae35u; s ^= s >> 16;
    float r = (float)(s >> 8) * 5.9604644775390625e-08f   // * 2^-24
            + 2.9802322387695312e-08f;                    // + 2^-25
    return -__builtin_amdgcn_logf(r);
}

__device__ __forceinline__ float softplus2(float v) {
    // softplus(v*ln2)/ln2 = max(v,0) + log2(1 + 2^-|v|): one v_exp + one v_log
    float e = __builtin_amdgcn_exp2f(-fabsf(v));
    return fmaxf(v, 0.0f) + __builtin_amdgcn_logf(1.0f + e);
}

__global__ void lq_zero_ws(float* __restrict__ ws) {
    if (threadIdx.x < 64) ws[threadIdx.x] = 0.0f;
}

__global__ __launch_bounds__(256) void lq_main(
        const float* __restrict__ x,
        const float* __restrict__ alpha,
        const float* __restrict__ dev,
        float* __restrict__ out,
        float* __restrict__ mean_ws,
        int nx)
{
    __shared__ float lmean[64];
    const int tid = threadIdx.x;
    if (tid < 64) lmean[tid] = 0.0f;
    __syncthreads();

    const int lane = tid & 31;
    const int row  = lane & 15;   // element slot in group == A-matrix row id
    const int hi   = lane >> 4;   // 0: bins 0-7, 1: bins 8-15
    const int wavesPerBlock = blockDim.x >> 5;
    const int wid     = blockIdx.x * wavesPerBlock + (tid >> 5);
    const int wstride = gridDim.x * wavesPerBlock;
    const int ngroups = (nx + 15) >> 4;

    // Constant A matrix: row0 = grid base (k-8.5), row1 = ones, else 0.
    // Lane holds K = hi*8 .. hi*8+7 in v16h elements 0..7; elements 8..15 (K>=16) = 0.
    v16h amat = {};
    if (row == 0) {
        #pragma unroll
        for (int j = 0; j < 8; ++j)
            amat[j] = (_Float16)((float)(hi * 8 + j) - 8.5f);
    } else if (row == 1) {
        #pragma unroll
        for (int j = 0; j < 8; ++j)
            amat[j] = (_Float16)1.0f;
    }

    for (int g = wid; g < ngroups; g += wstride) {
        const int  e     = g * 16 + row;
        const bool valid = (e < nx);
        const int  ec    = valid ? e : (nx - 1);

        // prefetch next tile for this wave (global_prefetch_b8)
        __builtin_prefetch(&x[(g + wstride < ngroups) ? (ec + (wstride << 4)) : ec], 0, 1);

        const float xv  = x[ec];
        const int   p   = ec & 63;
        const float al  = alpha[p];
        const float idv = __builtin_amdgcn_rcpf(dev[p]) * LOG2E_F;  // fold log2e
        const float a2  = al * idv;          // t2_k = a2*(k-8.5) - (x/dev)*log2e
        const float b2  = -xv * idv;
        const int   koff = hi << 3;

        // s = softplus2(-t2) tracked positively; cdf2 = -s
        float sprev = softplus2(-fmaf(a2, (float)koff - 8.5f, b2));
        const unsigned seedbase = ((unsigned)ec * 16u + (unsigned)koff) ^ 0x9e3779b9u;

        v16h bmat = {};                      // elements 8..15 (K>=16) stay zero
        #pragma unroll
        for (int j = 0; j < 8; ++j) {
            float snext = softplus2(-fmaf(a2, (float)(koff + j) - 7.5f, b2));
            float num2  = sprev - snext + NOISE2_C;
            float nlr2  = neg_log2_urand(seedbase + (unsigned)j);
            float w     = __builtin_amdgcn_sqrtf(num2)     // v_sqrt_f32
                        * __builtin_amdgcn_rsqf(nlr2);     // v_rsq_f32
            bmat[j]     = (_Float16)w;
            sprev       = snext;
        }

        // D[0][n] = sum_k base_k*w_k ; D[1][n] = sum_k w_k  (lands in lane n)
        v8f c = {};
        c = __builtin_amdgcn_wmma_f32_16x16x32_f16(
                /*neg_a=*/false, amat, /*neg_b=*/false, bmat,
                /*c_mod=*/(short)0, c, /*reuse_a=*/false, /*reuse_b=*/false);

        if (hi == 0 && valid) {
            out[e] = al * c[0] * __builtin_amdgcn_rcpf(c[1]);          // 1 v_rcp
            atomicAdd(&lmean[p], fabsf(xv * __builtin_amdgcn_rcpf(al)));
        }
    }

    __syncthreads();
    if (tid < 64) atomicAdd(&mean_ws[tid], lmean[tid]);
}

__global__ void lq_finalize(const float* __restrict__ ws,
                            float* __restrict__ out_tail, int cnt) {
    int i = threadIdx.x;
    if (i < 64) out_tail[i] = ws[i] * __builtin_amdgcn_rcpf((float)cnt);  // mean (8,8)
    if (i == 64) out_tail[64] = 0.0f;                                     // nzeros
}

extern "C" void kernel_launch(void* const* d_in, const int* in_sizes, int n_in,
                              void* d_out, int out_size, void* d_ws, size_t ws_size,
                              hipStream_t stream) {
    (void)n_in; (void)out_size; (void)ws_size;
    const float* x     = (const float*)d_in[0];
    const float* alpha = (const float*)d_in[1];
    const float* dev   = (const float*)d_in[2];
    const int    nx    = in_sizes[0];
    float*       out   = (float*)d_out;
    float*       ws    = (float*)d_ws;

    lq_zero_ws<<<1, 64, 0, stream>>>(ws);
    lq_main<<<2048, 256, 0, stream>>>(x, alpha, dev, out, ws, nx);
    lq_finalize<<<1, 128, 0, stream>>>(ws, out + nx, nx / 64);
}
